// PCTransformerEn_56581899157627
// MI455X (gfx1250) — matrix operand order repaired and checked by reading, not compiled, not run on hardware
//
#include <hip/hip_runtime.h>
#include <hip/hip_bf16.h>
#include <math.h>

// ---------------- model constants ----------------
#define BB    32
#define NN0   2048
#define KNN   16
#define NP0   512
#define NP1   128
#define DD    384
#define HH    6
#define HDIM  64
#define TDEPTH 6
#define HIDD  1536
#define MT    (BB * NP1)   // 4096 transformer rows

typedef __attribute__((ext_vector_type(16))) _Float16 v16h;
typedef __attribute__((ext_vector_type(8)))  _Float16 v8h;
typedef __attribute__((ext_vector_type(8)))  float    v8f;

union ABfrag { v16h v; v8h h[2]; };

__device__ __forceinline__ float gelu_exact(float x) {
    return 0.5f * x * (1.0f + erff(x * 0.70710678118654752f));
}

// =====================================================================
// Generic NT GEMM: C[M,N] = act(A[M,K] @ W[N,K]^T + bias) (+ residual)
// A, W are f16 row-major (K multiple of 32). Each wave computes a
// 32x32 output tile as 2x2 WMMA fragments (4 accumulators), reusing
// each A/B fragment twice -> 2 b128 loads per v_wmma instead of 4.
// Block = 128 threads = 4 waves arranged 2x2 over a 64x64 block tile.
// Batched via blockIdx.z with element strides.
// Fragment layout per CDNA5 ISA 7.12.2:
//   A (16x32 f16): lane L holds row M=L%16; VGPR0-3 = K[(L/16)*8 .. +7],
//                  VGPR4-7 = K[16+(L/16)*8 .. +7]  -> two b128 loads.
//   B (32x16 f16): lane L holds col N=L%16; VGPR0-3 = K[(L/16)*8 .. +7],
//                  VGPR4-7 = K[16+(L/16)*8 .. +7]  -> two b128 loads.
//   D (16x16 f32): lane L col N=L%16; VGPR j row M=(L/16)*8+j.
// =====================================================================
__global__ void gemm_nt_wmma(const _Float16* __restrict__ A,
                             const _Float16* __restrict__ W,
                             const float* __restrict__ bias,
                             const float* __restrict__ resid,
                             float* __restrict__ c32,
                             _Float16* __restrict__ c16,
                             int M, int N, int K,
                             long long aBS, long long wBS, long long cBS,
                             int act)
{
    int wave = threadIdx.x >> 5;
    int lane = threadIdx.x & 31;
    int wm = wave >> 1;
    int wn = wave & 1;
    int m0 = blockIdx.x * 64 + wm * 32;
    int n0 = blockIdx.y * 64 + wn * 32;
    if (m0 >= M || n0 >= N) return;
    long long z = blockIdx.z;
    int r  = lane & 15;
    int hi = lane >> 4;
    const _Float16* a0 = A + z * aBS + (long long)(m0 + r) * K;
    const _Float16* a1 = a0 + 16LL * K;
    const _Float16* w0 = W + z * wBS + (long long)(n0 + r) * K;
    const _Float16* w1 = w0 + 16LL * K;

    v8f c00 = {}, c01 = {}, c10 = {}, c11 = {};
    for (int k0 = 0; k0 < K; k0 += 32) {
        ABfrag fa0, fa1, fb0, fb1;
        fa0.h[0] = *(const v8h*)(a0 + k0 + hi * 8);
        fa0.h[1] = *(const v8h*)(a0 + k0 + 16 + hi * 8);
        fa1.h[0] = *(const v8h*)(a1 + k0 + hi * 8);
        fa1.h[1] = *(const v8h*)(a1 + k0 + 16 + hi * 8);
        fb0.h[0] = *(const v8h*)(w0 + k0 + hi * 8);
        fb0.h[1] = *(const v8h*)(w0 + k0 + 16 + hi * 8);
        fb1.h[0] = *(const v8h*)(w1 + k0 + hi * 8);
        fb1.h[1] = *(const v8h*)(w1 + k0 + 16 + hi * 8);
        c00 = __builtin_amdgcn_wmma_f32_16x16x32_f16(false, fa0.v, false, fb0.v, (short)0, c00, false, false);
        c01 = __builtin_amdgcn_wmma_f32_16x16x32_f16(false, fa0.v, false, fb1.v, (short)0, c01, false, false);
        c10 = __builtin_amdgcn_wmma_f32_16x16x32_f16(false, fa1.v, false, fb0.v, (short)0, c10, false, false);
        c11 = __builtin_amdgcn_wmma_f32_16x16x32_f16(false, fa1.v, false, fb1.v, (short)0, c11, false, false);
    }

    // write out the four 16x16 fragments
    #pragma unroll
    for (int tm = 0; tm < 2; ++tm) {
        #pragma unroll
        for (int tn = 0; tn < 2; ++tn) {
            v8f acc = (tm == 0) ? (tn == 0 ? c00 : c01) : (tn == 0 ? c10 : c11);
            int cn = n0 + tn * 16 + r;
            int mb = m0 + tm * 16 + hi * 8;
            float bv = bias ? bias[cn] : 0.0f;
            #pragma unroll
            for (int j = 0; j < 8; ++j) {
                float v = acc[j] + bv;
                if (act == 1) v = gelu_exact(v);
                long long ci = z * cBS + (long long)(mb + j) * N + cn;
                if (resid) v += resid[ci];
                if (c32) c32[ci] = v;
                else     c16[ci] = (_Float16)v;
            }
        }
    }
}

// ---------------- K=3 dense layer (input transform / pe layer 1) -----
__global__ void lin3_kernel(const float* __restrict__ x, const float* __restrict__ w,
                            const float* __restrict__ b, float* __restrict__ o32,
                            _Float16* __restrict__ o16, long long rows, int Cout, int act)
{
    long long t = blockIdx.x * (long long)blockDim.x + threadIdx.x;
    if (t >= rows * Cout) return;
    int o = (int)(t % Cout);
    long long r = t / Cout;
    const float* xr = x + r * 3;
    float v = b[o] + xr[0] * w[o * 3] + xr[1] * w[o * 3 + 1] + xr[2] * w[o * 3 + 2];
    if (act) v = gelu_exact(v);
    if (o32) o32[t] = v;
    else     o16[t] = (_Float16)v;
}

// ---------------- KNN (top-16 smallest squared distance) -------------
__global__ void knn_kernel(const float* __restrict__ ref, const float* __restrict__ qry,
                           int* __restrict__ idx, int Nq, int Nk)
{
    int t = blockIdx.x * blockDim.x + threadIdx.x;
    if (t >= BB * Nq) return;
    int b = t / Nq;
    const float* rb = ref + (long long)b * Nk * 3;
    float qx = qry[(long long)t * 3 + 0];
    float qy = qry[(long long)t * 3 + 1];
    float qz = qry[(long long)t * 3 + 2];
    float bd[KNN]; int bi[KNN];
    #pragma unroll
    for (int j = 0; j < KNN; ++j) { bd[j] = 3.0e38f; bi[j] = 0; }
    for (int j = 0; j < Nk; ++j) {
        float dx = rb[j * 3 + 0] - qx;
        float dy = rb[j * 3 + 1] - qy;
        float dz = rb[j * 3 + 2] - qz;
        float dd = dx * dx + dy * dy + dz * dz;
        if (dd < bd[KNN - 1]) {
            int p = KNN - 1;
            while (p > 0 && bd[p - 1] > dd) {
                bd[p] = bd[p - 1]; bi[p] = bi[p - 1]; --p;
            }
            bd[p] = dd; bi[p] = j;
        }
    }
    #pragma unroll
    for (int j = 0; j < KNN; ++j) idx[(long long)t * KNN + j] = bi[j];
}

// ---------------- farthest point sampling (1 block per batch) --------
__global__ void fps_kernel(const float* __restrict__ xyz, float* __restrict__ dbuf,
                           int* __restrict__ idx, int Np, int npoint)
{
    __shared__ float sred[256];
    __shared__ int   sidx[256];
    __shared__ int   slast;
    int b = blockIdx.x;
    int tid = threadIdx.x;
    const float* x = xyz + (long long)b * Np * 3;
    float* d = dbuf + (long long)b * Np;
    for (int j = tid; j < Np; j += 256) d[j] = 1.0e10f;
    if (tid == 0) { idx[b * npoint] = 0; slast = 0; }
    __syncthreads();
    for (int it = 1; it < npoint; ++it) {
        int last = slast;
        float px = x[last * 3], py = x[last * 3 + 1], pz = x[last * 3 + 2];
        float bestd = -1.0f; int besti = 0;
        for (int j = tid; j < Np; j += 256) {
            float dx = x[j * 3] - px, dy = x[j * 3 + 1] - py, dz = x[j * 3 + 2] - pz;
            float nd = fminf(d[j], dx * dx + dy * dy + dz * dz);
            d[j] = nd;
            if (nd > bestd) { bestd = nd; besti = j; }
        }
        sred[tid] = bestd; sidx[tid] = besti;
        __syncthreads();
        for (int s = 128; s > 0; s >>= 1) {
            if (tid < s) {
                if (sred[tid + s] > sred[tid] ||
                    (sred[tid + s] == sred[tid] && sidx[tid + s] < sidx[tid])) {
                    sred[tid] = sred[tid + s]; sidx[tid] = sidx[tid + s];
                }
            }
            __syncthreads();
        }
        if (tid == 0) { idx[b * npoint + it] = sidx[0]; slast = sidx[0]; }
        __syncthreads();
    }
}

// ---------------- gather rows by index -------------------------------
__global__ void gather_kernel(const float* __restrict__ src, const int* __restrict__ idx,
                              float* __restrict__ dst, int Nq, int Nk, int C)
{
    long long t = blockIdx.x * (long long)blockDim.x + threadIdx.x;
    long long tot = (long long)BB * Nq * C;
    if (t >= tot) return;
    int c = (int)(t % C);
    long long r = t / C;
    int q = (int)(r % Nq);
    int b = (int)(r / Nq);
    dst[t] = src[((long long)b * Nk + idx[b * Nq + q]) * C + c];
}

// ---------------- edge-feature build: [nbr - fq, fq] -> f16 ----------
__global__ void build_feat(const float* __restrict__ fk, const float* __restrict__ fq,
                           const int* __restrict__ idx, _Float16* __restrict__ feat,
                           int Nq, int Nk, int Cf, int Kpad)
{
    long long t = blockIdx.x * (long long)blockDim.x + threadIdx.x;
    long long tot = (long long)BB * Nq * KNN * Cf;
    if (t >= tot) return;
    int c = (int)(t % Cf);
    long long rest = t / Cf;
    int kk = (int)(rest % KNN);
    long long bq = rest / KNN;           // b*Nq + q
    int b = (int)(bq / Nq);
    int nb = idx[bq * KNN + kk];
    float vq = fq[bq * Cf + c];
    float vn = fk[((long long)b * Nk + nb) * Cf + c];
    long long row = bq * KNN + kk;
    feat[row * Kpad + c]      = (_Float16)(vn - vq);
    feat[row * Kpad + Cf + c] = (_Float16)vq;
    for (int zpos = 2 * Cf + c; zpos < Kpad; zpos += Cf)
        feat[row * Kpad + zpos] = (_Float16)0.0f;
}

// ---------------- group-norm stats (per batch x group of 4) ----------
__global__ void gn_stats(const _Float16* __restrict__ h, float* __restrict__ stats,
                         int Nq, int C, int cg)
{
    __shared__ float ssum[256], ssq[256];
    int b = blockIdx.x >> 2;
    int g = blockIdx.x & 3;
    long long base = (long long)b * Nq * KNN * C + g * cg;
    int total = Nq * KNN * cg;
    float s = 0.f, sq = 0.f;
    for (int i = threadIdx.x; i < total; i += 256) {
        int c  = i % cg;
        int rk = i / cg;
        float v = (float)h[base + (long long)rk * C + c];
        s += v; sq += v * v;
    }
    ssum[threadIdx.x] = s; ssq[threadIdx.x] = sq;
    __syncthreads();
    for (int st = 128; st > 0; st >>= 1) {
        if (threadIdx.x < st) {
            ssum[threadIdx.x] += ssum[threadIdx.x + st];
            ssq[threadIdx.x]  += ssq[threadIdx.x + st];
        }
        __syncthreads();
    }
    if (threadIdx.x == 0) {
        float inv = 1.0f / (float)total;
        float m = ssum[0] * inv;
        stats[blockIdx.x * 2]     = m;
        stats[blockIdx.x * 2 + 1] = ssq[0] * inv - m * m;
    }
}

// ---------------- normalize + leaky-relu + max over k ----------------
__global__ void gn_lrelu_max(const _Float16* __restrict__ h, const float* __restrict__ stats,
                             const float* __restrict__ gamma, const float* __restrict__ beta,
                             float* __restrict__ fout, int Nq, int C, int cg)
{
    long long t = blockIdx.x * (long long)blockDim.x + threadIdx.x;
    long long tot = (long long)BB * Nq * C;
    if (t >= tot) return;
    int o = (int)(t % C);
    long long r = t / C;               // b*Nq + q
    int b = (int)(r / Nq);
    int g = o / cg;
    float m   = stats[(b * 4 + g) * 2];
    float inv = rsqrtf(stats[(b * 4 + g) * 2 + 1] + 1e-5f);
    float gm = gamma[o], bt = beta[o];
    long long base = r * KNN * C + o;
    float best = -3.0e38f;
    #pragma unroll
    for (int kk = 0; kk < KNN; ++kk) {
        float v = ((float)h[base + (long long)kk * C] - m) * inv * gm + bt;
        v = (v > 0.f) ? v : 0.2f * v;
        best = fmaxf(best, v);
    }
    fout[t] = best;
}

// ---------------- layernorm (wave32 per row) -> f16 ------------------
__global__ void layernorm_f16(const float* __restrict__ x, const float* __restrict__ g,
                              const float* __restrict__ b, _Float16* __restrict__ out,
                              int rows, int C)
{
    int wave = threadIdx.x >> 5;
    int lane = threadIdx.x & 31;
    int row = blockIdx.x * 4 + wave;
    if (row >= rows) return;
    const float* xr = x + (long long)row * C;
    float s = 0.f, sq = 0.f;
    for (int c = lane; c < C; c += 32) { float v = xr[c]; s += v; sq += v * v; }
    #pragma unroll
    for (int off = 16; off > 0; off >>= 1) {
        s  += __shfl_down(s, off, 32);
        sq += __shfl_down(sq, off, 32);
    }
    s = __shfl(s, 0, 32); sq = __shfl(sq, 0, 32);
    float m = s / C;
    float inv = rsqrtf(sq / C - m * m + 1e-5f);
    for (int c = lane; c < C; c += 32)
        out[(long long)row * C + c] = (_Float16)((xr[c] - m) * inv * g[c] + b[c]);
}

// ---------------- softmax rows (wave32 per row) f32 -> f16 -----------
__global__ void softmax_f16(const float* __restrict__ a, _Float16* __restrict__ out,
                            int rows, int C)
{
    int wave = threadIdx.x >> 5;
    int lane = threadIdx.x & 31;
    int row = blockIdx.x * 4 + wave;
    if (row >= rows) return;
    const float* ar = a + (long long)row * C;
    float mx = -3.0e38f;
    for (int c = lane; c < C; c += 32) mx = fmaxf(mx, ar[c]);
    #pragma unroll
    for (int off = 16; off > 0; off >>= 1) mx = fmaxf(mx, __shfl_down(mx, off, 32));
    mx = __shfl(mx, 0, 32);
    float s = 0.f;
    for (int c = lane; c < C; c += 32) s += expf(ar[c] - mx);
    #pragma unroll
    for (int off = 16; off > 0; off >>= 1) s += __shfl_down(s, off, 32);
    s = __shfl(s, 0, 32);
    float inv = 1.0f / s;
    for (int c = lane; c < C; c += 32)
        out[(long long)row * C + c] = (_Float16)(expf(ar[c] - mx) * inv);
}

// ---------------- qkv split + scale + V transpose --------------------
__global__ void qkv_split(const _Float16* __restrict__ qkv, _Float16* __restrict__ q,
                          _Float16* __restrict__ k, _Float16* __restrict__ vT)
{
    long long t = blockIdx.x * (long long)blockDim.x + threadIdx.x;
    long long tot = (long long)BB * HH * NP1 * HDIM;
    if (t >= tot) return;
    int d = (int)(t % HDIM);
    int s = (int)((t / HDIM) % NP1);
    int h = (int)((t / (HDIM * NP1)) % HH);
    int b = (int)(t / ((long long)HDIM * NP1 * HH));
    long long row = ((long long)(b * NP1 + s)) * (3 * DD) + h * HDIM + d;
    long long bh = b * HH + h;
    q[(bh * NP1 + s) * HDIM + d]  = (_Float16)((float)qkv[row] * 0.125f);
    k[(bh * NP1 + s) * HDIM + d]  = qkv[row + DD];
    vT[(bh * HDIM + d) * NP1 + s] = qkv[row + 2 * DD];
}

// ---------------- merge heads: [B,H,S,HD] -> [B*S, D] ----------------
__global__ void o_merge(const _Float16* __restrict__ o, _Float16* __restrict__ out)
{
    long long t = blockIdx.x * (long long)blockDim.x + threadIdx.x;
    long long tot = (long long)MT * DD;
    if (t >= tot) return;
    int c = (int)(t % DD);
    int s = (int)((t / DD) % NP1);
    int b = (int)(t / ((long long)DD * NP1));
    int h = c / HDIM, d = c % HDIM;
    out[t] = o[(((long long)(b * HH + h) * NP1) + s) * HDIM + d];
}

// ---------------- misc elementwise ----------------------------------
__global__ void add_f32(float* __restrict__ x, const float* __restrict__ y, long long n)
{
    long long t = blockIdx.x * (long long)blockDim.x + threadIdx.x;
    if (t < n) x[t] += y[t];
}
__global__ void copy_f32(const float* __restrict__ s, float* __restrict__ d, long long n)
{
    long long t = blockIdx.x * (long long)blockDim.x + threadIdx.x;
    if (t < n) d[t] = s[t];
}
__global__ void cvt_f16(const float* __restrict__ s, _Float16* __restrict__ d, long long n)
{
    long long t = blockIdx.x * (long long)blockDim.x + threadIdx.x;
    if (t < n) d[t] = (_Float16)s[t];
}
__global__ void cvt_w_pad(const float* __restrict__ w, _Float16* __restrict__ wp,
                          int N, int K, int Kpad)
{
    long long t = blockIdx.x * (long long)blockDim.x + threadIdx.x;
    if (t >= (long long)N * Kpad) return;
    int k = (int)(t % Kpad);
    int n = (int)(t / Kpad);
    wp[t] = (k < K) ? (_Float16)w[(long long)n * K + k] : (_Float16)0.0f;
}

// =====================================================================
extern "C" void kernel_launch(void* const* d_in, const int* in_sizes, int n_in,
                              void* d_out, int out_size, void* d_ws, size_t ws_size,
                              hipStream_t stream)
{
    (void)in_sizes; (void)n_in; (void)out_size; (void)ws_size;
    const float* xyz   = (const float*)d_in[0];
    const float* it_w  = (const float*)d_in[1];
    const float* it_b  = (const float*)d_in[2];
    const float* l1_w  = (const float*)d_in[3];
    const float* gn1_g = (const float*)d_in[4];
    const float* gn1_b = (const float*)d_in[5];
    const float* l2_w  = (const float*)d_in[6];
    const float* gn2_g = (const float*)d_in[7];
    const float* gn2_b = (const float*)d_in[8];
    const float* l3_w  = (const float*)d_in[9];
    const float* gn3_g = (const float*)d_in[10];
    const float* gn3_b = (const float*)d_in[11];
    const float* l4_w  = (const float*)d_in[12];
    const float* gn4_g = (const float*)d_in[13];
    const float* gn4_b = (const float*)d_in[14];
    const float* pe_w1 = (const float*)d_in[15];
    const float* pe_b1 = (const float*)d_in[16];
    const float* pe_w2 = (const float*)d_in[17];
    const float* pe_b2 = (const float*)d_in[18];
    const float* ip_w1 = (const float*)d_in[19];
    const float* ip_b1 = (const float*)d_in[20];
    const float* ip_w2 = (const float*)d_in[21];
    const float* ip_b2 = (const float*)d_in[22];
    const float* ln1_g = (const float*)d_in[23];
    const float* ln1_b = (const float*)d_in[24];
    const float* qkv_w = (const float*)d_in[25];
    const float* proj_w= (const float*)d_in[26];
    const float* proj_b= (const float*)d_in[27];
    const float* ln2_g = (const float*)d_in[28];
    const float* ln2_b = (const float*)d_in[29];
    const float* fc1_w = (const float*)d_in[30];
    const float* fc1_b = (const float*)d_in[31];
    const float* fc2_w = (const float*)d_in[32];
    const float* fc2_b = (const float*)d_in[33];

    char* ws = (char*)d_ws;
    size_t off = 0;
    auto alloc = [&](size_t bytes) -> char* {
        char* p = ws + off;
        off = (off + bytes + 255) & ~(size_t)255;
        return p;
    };
    auto allocH = [&](size_t n) -> _Float16* { return (_Float16*)alloc(n * 2); };
    auto allocF = [&](size_t n) -> float*    { return (float*)alloc(n * 4); };
    auto allocI = [&](size_t n) -> int*      { return (int*)alloc(n * 4); };

    // f16 weights
    _Float16* l1w16  = allocH(32 * 32);
    _Float16* l2w16  = allocH(64 * 64);
    _Float16* l3w16  = allocH(64 * 128);
    _Float16* l4w16  = allocH(128 * 128);
    _Float16* pew216 = allocH((size_t)DD * 128);
    _Float16* ipw116 = allocH((size_t)512 * 128);
    _Float16* ipw216 = allocH((size_t)DD * 512);
    _Float16* qkvw16 = allocH((size_t)TDEPTH * 3 * DD * DD);
    _Float16* projw16= allocH((size_t)TDEPTH * DD * DD);
    _Float16* fc1w16 = allocH((size_t)TDEPTH * HIDD * DD);
    _Float16* fc2w16 = allocH((size_t)TDEPTH * DD * HIDD);

    // activations / scratch
    float*    f0     = allocF((size_t)BB * NN0 * 8);
    int*      idxbuf = allocI((size_t)BB * NN0 * KNN);
    _Float16* feat   = allocH((size_t)33554432);      // shared feat arena (max 64MB)
    _Float16* harena = allocH((size_t)33554432);      // shared GEMM-out arena (f16)
    float*    stats  = allocF(256);
    float*    f1     = allocF((size_t)BB * NN0 * 32);
    float*    fpsd   = allocF((size_t)BB * NN0);
    int*      idx0   = allocI((size_t)BB * NP0);
    int*      idx1   = allocI((size_t)BB * NP1);
    float*    coorq1 = allocF((size_t)BB * NP0 * 3);
    float*    fq1    = allocF((size_t)BB * NP0 * 32);
    float*    f2     = allocF((size_t)BB * NP0 * 64);
    float*    f3     = allocF((size_t)BB * NP0 * 64);
    float*    coorq2 = allocF((size_t)BB * NP1 * 3);
    float*    fq2    = allocF((size_t)BB * NP1 * 64);
    float*    f4     = allocF((size_t)BB * NP1 * 128);
    _Float16* f4h    = allocH((size_t)MT * 128);
    _Float16* peh16  = allocH((size_t)MT * 128);
    float*    pe     = allocF((size_t)MT * DD);
    _Float16* iph16  = allocH((size_t)MT * 512);
    float*    x      = allocF((size_t)MT * DD);
    _Float16* h16    = allocH((size_t)MT * DD);
    _Float16* qkv16  = allocH((size_t)MT * 3 * DD);
    _Float16* q16    = allocH((size_t)BB * HH * NP1 * HDIM);
    _Float16* k16    = allocH((size_t)BB * HH * NP1 * HDIM);
    _Float16* vT16   = allocH((size_t)BB * HH * HDIM * NP1);
    float*    att    = allocF((size_t)BB * HH * NP1 * NP1);
    _Float16* attp16 = allocH((size_t)BB * HH * NP1 * NP1);
    _Float16* o16    = allocH((size_t)BB * HH * NP1 * HDIM);
    _Float16* of16   = allocH((size_t)MT * DD);

    auto GB = [](long long n, int bs) { return dim3((unsigned)((n + bs - 1) / bs)); };

    auto gemm = [&](const _Float16* A, const _Float16* Wm, const float* bias,
                    const float* resid, float* c32, _Float16* c16,
                    int M, int N, int K, int batches,
                    long long aBS, long long wBS, long long cBS, int act) {
        dim3 grid((M + 63) / 64, (N + 63) / 64, batches);
        gemm_nt_wmma<<<grid, 128, 0, stream>>>(A, Wm, bias, resid, c32, c16,
                                               M, N, K, aBS, wBS, cBS, act);
    };
    auto cvtw = [&](const float* w, _Float16* o, int N, int K, int Kpad) {
        long long t = (long long)N * Kpad;
        cvt_w_pad<<<GB(t, 256), 256, 0, stream>>>(w, o, N, K, Kpad);
    };

    // ---------------- weight conversion ----------------
    cvtw(l1_w, l1w16, 32, 16, 32);
    cvtw(l2_w, l2w16, 64, 64, 64);
    cvtw(l3_w, l3w16, 64, 128, 128);
    cvtw(l4_w, l4w16, 128, 128, 128);
    cvtw(pe_w2, pew216, DD, 128, 128);
    cvtw(ip_w1, ipw116, 512, 128, 128);
    cvtw(ip_w2, ipw216, DD, 512, 512);
    cvtw(qkv_w, qkvw16, TDEPTH * 3 * DD, DD, DD);
    cvtw(proj_w, projw16, TDEPTH * DD, DD, DD);
    cvtw(fc1_w, fc1w16, TDEPTH * HIDD, DD, DD);
    cvtw(fc2_w, fc2w16, TDEPTH * DD, HIDD, HIDD);

    // ---------------- graph layer helper ----------------
    auto graph_layer = [&](const float* cq, const float* ck, const float* fqv,
                           const float* fkv, int Nq, int Nk, int Cf,
                           const _Float16* w16, int Cout, int Kpad,
                           const float* g, const float* bt, float* fout) {
        knn_kernel<<<GB((long long)BB * Nq, 128), 128, 0, stream>>>(ck, cq, idxbuf, Nq, Nk);
        long long tf = (long long)BB * Nq * KNN * Cf;
        build_feat<<<GB(tf, 256), 256, 0, stream>>>(fkv, fqv, idxbuf, feat, Nq, Nk, Cf, Kpad);
        int M = BB * Nq * KNN;
        gemm(feat, w16, nullptr, nullptr, nullptr, harena, M, Cout, Kpad, 1, 0, 0, 0, 0);
        gn_stats<<<dim3(BB * 4), 256, 0, stream>>>(harena, stats, Nq, Cout, Cout / 4);
        long long tn = (long long)BB * Nq * Cout;
        gn_lrelu_max<<<GB(tn, 256), 256, 0, stream>>>(harena, stats, g, bt, fout,
                                                      Nq, Cout, Cout / 4);
    };

    // ---------------- backbone ----------------
    lin3_kernel<<<GB((long long)BB * NN0 * 8, 256), 256, 0, stream>>>(
        xyz, it_w, it_b, f0, nullptr, (long long)BB * NN0, 8, 0);

    graph_layer(xyz, xyz, f0, f0, NN0, NN0, 8,  l1w16, 32, 32, gn1_g, gn1_b, f1);

    fps_kernel<<<dim3(BB), 256, 0, stream>>>(xyz, fpsd, idx0, NN0, NP0);
    gather_kernel<<<GB((long long)BB * NP0 * 3, 256), 256, 0, stream>>>(xyz, idx0, coorq1, NP0, NN0, 3);
    gather_kernel<<<GB((long long)BB * NP0 * 32, 256), 256, 0, stream>>>(f1, idx0, fq1, NP0, NN0, 32);

    graph_layer(coorq1, xyz, fq1, f1, NP0, NN0, 32, l2w16, 64, 64, gn2_g, gn2_b, f2);
    graph_layer(coorq1, coorq1, f2, f2, NP0, NP0, 64, l3w16, 64, 128, gn3_g, gn3_b, f3);

    fps_kernel<<<dim3(BB), 256, 0, stream>>>(coorq1, fpsd, idx1, NP0, NP1);
    gather_kernel<<<GB((long long)BB * NP1 * 3, 256), 256, 0, stream>>>(coorq1, idx1, coorq2, NP1, NP0, 3);
    gather_kernel<<<GB((long long)BB * NP1 * 64, 256), 256, 0, stream>>>(f3, idx1, fq2, NP1, NP0, 64);

    graph_layer(coorq2, coorq1, fq2, f3, NP1, NP0, 64, l4w16, 128, 128, gn4_g, gn4_b, f4);

    // pe = gelu(coor @ pe_w1^T + b1) @ pe_w2^T + b2
    lin3_kernel<<<GB((long long)MT * 128, 256), 256, 0, stream>>>(
        coorq2, pe_w1, pe_b1, nullptr, peh16, (long long)MT, 128, 1);
    gemm(peh16, pew216, pe_b2, nullptr, pe, nullptr, MT, DD, 128, 1, 0, 0, 0, 0);

    // x = gelu(f4 @ ip_w1^T + b1) @ ip_w2^T + b2 + pe
    cvt_f16<<<GB((long long)MT * 128, 256), 256, 0, stream>>>(f4, f4h, (long long)MT * 128);
    gemm(f4h, ipw116, ip_b1, nullptr, nullptr, iph16, MT, 512, 128, 1, 0, 0, 0, 1);
    gemm(iph16, ipw216, ip_b2, nullptr, x, nullptr, MT, DD, 512, 1, 0, 0, 0, 0);
    add_f32<<<GB((long long)MT * DD, 256), 256, 0, stream>>>(x, pe, (long long)MT * DD);

    // ---------------- transformer ----------------
    const int BHS = BB * HH;  // 192 batched attention mats
    for (int i = 0; i < TDEPTH; ++i) {
        layernorm_f16<<<GB(MT, 4), 128, 0, stream>>>(x, ln1_g + i * DD, ln1_b + i * DD, h16, MT, DD);
        gemm(h16, qkvw16 + (size_t)i * 3 * DD * DD, nullptr, nullptr, nullptr, qkv16,
             MT, 3 * DD, DD, 1, 0, 0, 0, 0);
        qkv_split<<<GB((long long)BHS * NP1 * HDIM, 256), 256, 0, stream>>>(qkv16, q16, k16, vT16);
        // att = (q*scale) @ k^T : batched
        gemm(q16, k16, nullptr, nullptr, att, nullptr, NP1, NP1, HDIM, BHS,
             (long long)NP1 * HDIM, (long long)NP1 * HDIM, (long long)NP1 * NP1, 0);
        softmax_f16<<<GB((long long)BHS * NP1, 4), 128, 0, stream>>>(att, attp16, BHS * NP1, NP1);
        // o = att @ v  (v stored transposed [HD,S])
        gemm(attp16, vT16, nullptr, nullptr, nullptr, o16, NP1, HDIM, NP1, BHS,
             (long long)NP1 * NP1, (long long)HDIM * NP1, (long long)NP1 * HDIM, 0);
        o_merge<<<GB((long long)MT * DD, 256), 256, 0, stream>>>(o16, of16);
        // x = x + o @ proj^T + proj_b
        gemm(of16, projw16 + (size_t)i * DD * DD, proj_b + i * DD, x, x, nullptr,
             MT, DD, DD, 1, 0, 0, 0, 0);
        // mlp
        layernorm_f16<<<GB(MT, 4), 128, 0, stream>>>(x, ln2_g + i * DD, ln2_b + i * DD, h16, MT, DD);
        gemm(h16, fc1w16 + (size_t)i * HIDD * DD, fc1_b + i * HIDD, nullptr, nullptr,
             harena /*reuse as f16 hid*/, MT, HIDD, DD, 1, 0, 0, 0, 1);
        gemm(harena, fc2w16 + (size_t)i * DD * HIDD, fc2_b + i * DD, x, x, nullptr,
             MT, DD, HIDD, 1, 0, 0, 0, 0);
    }

    // ---------------- outputs: (coor, x) concatenated ----------------
    float* outp = (float*)d_out;
    copy_f32<<<GB((long long)BB * NP1 * 3, 256), 256, 0, stream>>>(coorq2, outp, (long long)BB * NP1 * 3);
    copy_f32<<<GB((long long)MT * DD, 256), 256, 0, stream>>>(x, outp + (long long)BB * NP1 * 3, (long long)MT * DD);
}
